// SessionSpatialProjector_20882130993235
// MI455X (gfx1250) — compile-verified, matrix operand order checked by reading, not run on hardware
//
#include <hip/hip_runtime.h>
#include <hip/hip_bf16.h>
#include <math.h>

// ---------------------------------------------------------------------------
// SessionSpatialProjector fused kernel for gfx1250 (MI455X)
//   pre-pass: W_sess, W2 -> pre-transposed bf16 in d_ws (L2-resident, 2.25MB)
//   main:     h   = gelu(W^T x + b)   (K=256 -> M=512)   bf16 WMMA
//             out = (W2^T h + b2)*tm  (K=512 -> M=256)   bf16 WMMA
// 8 wave32 per workgroup, one (sample, 32-time-col) tile, h stays in LDS.
// Weight tiles double-buffered via GLOBAL_LOAD_ASYNC_TO_LDS (ASYNCcnt).
// Activations carried as bf16 hi + lo residual => only weight-rounding error.
// ---------------------------------------------------------------------------

typedef __bf16 bf16_t;
typedef bf16_t bf16x16 __attribute__((ext_vector_type(16)));
typedef float  f32x8   __attribute__((ext_vector_type(8)));
typedef int    v4i     __attribute__((ext_vector_type(4)));

#define MAXC    256
#define MAXT    4096
#define HID     512
#define NSRC    256

#define NT      32            // time columns per workgroup
#define XROW    264           // 256 + 8 halfs pad   (528 B, 16B aligned)
#define WROW    48            // 32 + 16 halfs pad   (96 B, 16B aligned)
#define HROW    520           // 512 + 8 halfs pad   (1040 B, 16B aligned)

#define XHI_OFF 0
#define XLO_OFF (NT * XROW)                  // 8448
#define WT_OFF  (2 * NT * XROW)              // 16896
#define WTBUF   (HID * WROW)                 // 24576 halfs per buffer
#define HHI_OFF (WT_OFF + 2 * WTBUF)         // 66048
#define HLO_OFF (HHI_OFF + NT * HROW)        // 82688
#define SMEM_HALFS (HLO_OFF + NT * HROW)     // 99328 halfs = 198656 B (<320KB)

// workspace layout (bf16 halfs): Wt1[sid][h][c] then Wt2[src][h]
#define WSESS_ELEMS (8 * MAXC * HID)         // 1048576
#define W2_ELEMS    (HID * NSRC)             // 131072
#define WS2_OFF     WSESS_ELEMS

static __device__ __forceinline__ unsigned short f2bf(float f) {
    return __builtin_bit_cast(unsigned short, (bf16_t)f);   // native cvt, RTNE
}
static __device__ __forceinline__ float bf2f(unsigned short h) {
    return __uint_as_float((unsigned)h << 16);
}

// ---- async global->LDS copy path (gfx1250), guarded with sync fallback ----
#if defined(__AMDGCN__) && __has_builtin(__builtin_amdgcn_global_load_async_to_lds_b128)
#define HAVE_ASYNC_LDS 1
typedef v4i __attribute__((address_space(1)))* gv4i_p;   // global int4*
typedef v4i __attribute__((address_space(3)))* lv4i_p;   // LDS int4*
static __device__ __forceinline__ void cp16(const unsigned short* g, unsigned short* l) {
    __builtin_amdgcn_global_load_async_to_lds_b128(
        (gv4i_p)(unsigned short*)g,   // drop const, addrspacecast to AS1
        (lv4i_p)l,                    // addrspacecast generic->AS3
        0, 0);
}
#if __has_builtin(__builtin_amdgcn_s_wait_asynccnt)
#define WAIT_ASYNC_8() __builtin_amdgcn_s_wait_asynccnt(8)
#define WAIT_ASYNC_4() __builtin_amdgcn_s_wait_asynccnt(4)
#define WAIT_ASYNC_0() __builtin_amdgcn_s_wait_asynccnt(0)
#else
#define WAIT_ASYNC_8() asm volatile("s_wait_asynccnt 8" ::: "memory")
#define WAIT_ASYNC_4() asm volatile("s_wait_asynccnt 4" ::: "memory")
#define WAIT_ASYNC_0() asm volatile("s_wait_asynccnt 0" ::: "memory")
#endif
#else
static __device__ __forceinline__ void cp16(const unsigned short* g, unsigned short* l) {
    *reinterpret_cast<uint4*>(l) = *reinterpret_cast<const uint4*>(g);
}
#define WAIT_ASYNC_8() ((void)0)
#define WAIT_ASYNC_4() ((void)0)
#define WAIT_ASYNC_0() ((void)0)
#endif

// Fragment layout for 16-bit WMMA A/B operands (CDNA5 ISA 7.12.2):
// lane half s=0 holds K={0..7,16..23}, s=1 holds K={8..15,24..31}:
// two contiguous 8-half runs 32 B apart -> two ds_load_b128.
static __device__ __forceinline__ bf16x16 load_frag(const unsigned short* p) {
    union { uint4 q[2]; bf16x16 v; } u;
    u.q[0] = *reinterpret_cast<const uint4*>(p);
    u.q[1] = *reinterpret_cast<const uint4*>(p + 16);
    return u.v;
}

// ---------------- pre-pass: fp32 weights -> transposed bf16 in ws ----------
__global__ __launch_bounds__(256) void convert_weights_kernel(
    const float* __restrict__ W_sess,   // (8, 256, 512)  [sid][c][h]
    const float* __restrict__ W2,       // (512, 256)     [h][s]
    unsigned short* __restrict__ wsb)
{
    int i = blockIdx.x * 256 + threadIdx.x;
    if (i < WSESS_ELEMS) {              // coalesced read along h
        int h = i & 511, c = (i >> 9) & 255, sid = i >> 17;
        wsb[(((sid << 9) + h) << 8) + c] = f2bf(W_sess[i]);   // Wt1[sid][h][c]
    } else {
        int j = i - WSESS_ELEMS;
        if (j < W2_ELEMS) {             // coalesced read along s
            int sD = j & 255, h = j >> 8;
            wsb[WS2_OFF + (sD << 9) + h] = f2bf(W2[j]);       // Wt2[s][h]
        }
    }
}

// ------------------------------- main kernel -------------------------------
__global__ __launch_bounds__(256) void ssp_fused_kernel(
    const float* __restrict__ x,
    const int*   __restrict__ session_ids,
    const int*   __restrict__ channel_counts,
    const int*   __restrict__ seq_lens,
    const float* __restrict__ b_sess,   // (8, 512)
    const float* __restrict__ b2,       // (256,)
    const unsigned short* __restrict__ wsb,  // bf16 weights (pre-pass)
    float*       __restrict__ out)
{
    __shared__ __align__(16) unsigned short smem[SMEM_HALFS];
    unsigned short* xhi = smem + XHI_OFF;
    unsigned short* xlo = smem + XLO_OFF;
    unsigned short* hhi = smem + HHI_OFF;
    unsigned short* hlo = smem + HLO_OFF;

    const int tid  = threadIdx.x;
    const int wave = tid >> 5;            // wave32
    const int lane = tid & 31;
    const int lh   = lane & 15;
    const int s    = lane >> 4;

    const int nblk = (int)blockIdx.x & 127;
    const int b    = (int)blockIdx.x >> 7;
    const int t0   = nblk * NT;

    const int sid  = session_ids[b];
    const int cc   = channel_counts[b];
    const int slen = seq_lens[b];

    const unsigned short* Wt1 = wsb + (size_t)sid * HID * MAXC;  // [h][c]
    const unsigned short* Wt2 = wsb + WS2_OFF;                   // [s][h]

    // ---- kick off async stage of W chunk 0 into weight buffer 0 ----
    {
        unsigned short* wb = smem + WT_OFF;
        for (int e = tid; e < HID * 4; e += 256) {     // 512 rows x 4 16B segs
            int h = e >> 2, seg = e & 3;
            cp16(Wt1 + (size_t)h * MAXC + seg * 8, wb + h * WROW + seg * 8);
        }
    }

    // ---- load + transpose x tile into LDS bf16 hi/lo: xT[t][c] (overlaps) --
    {
        const float* xb = x + ((size_t)b * MAXC) * MAXT + t0;
        for (int e = tid; e < MAXC * (NT / 4); e += 256) {
            int c  = e >> 3;
            int t4 = (e & 7) * 4;
            float4 v = {0.f, 0.f, 0.f, 0.f};
            if (c < cc)
                v = *reinterpret_cast<const float4*>(xb + (size_t)c * MAXT + t4);
            const float vv[4] = {v.x, v.y, v.z, v.w};
            #pragma unroll
            for (int j = 0; j < 4; ++j) {
                unsigned short hi = f2bf(vv[j]);
                xhi[(t4 + j) * XROW + c] = hi;
                xlo[(t4 + j) * XROW + c] = f2bf(vv[j] - bf2f(hi));
            }
        }
    }

    // ---- stage 1: H = gelu(W^T x + b), tile 512 x NT, double-buffered W ----
    f32x8 acc[4][2];
    {
        const f32x8 z = {0.f, 0.f, 0.f, 0.f, 0.f, 0.f, 0.f, 0.f};
        #pragma unroll
        for (int mi = 0; mi < 4; ++mi)
            #pragma unroll
            for (int ni = 0; ni < 2; ++ni) acc[mi][ni] = z;
    }
    const int mbase = wave * 64;

    for (int ks = 0; ks < 8; ++ks) {
        if (ks < 7) {        // prefetch next W chunk into other buffer
            unsigned short* wb = smem + WT_OFF + ((ks + 1) & 1) * WTBUF;
            const unsigned short* g0 = Wt1 + (ks + 1) * 32;
            for (int e = tid; e < HID * 4; e += 256) {
                int h = e >> 2, seg = e & 3;
                cp16(g0 + (size_t)h * MAXC + seg * 8, wb + h * WROW + seg * 8);
            }
            WAIT_ASYNC_8();
        } else {             // prefetch stage-2 W2 chunk 0 into buffer 0
            unsigned short* wb = smem + WT_OFF;
            for (int e = tid; e < NSRC * 4; e += 256) {
                int r = e >> 2, seg = e & 3;
                cp16(Wt2 + (size_t)r * HID + seg * 8, wb + r * WROW + seg * 8);
            }
            WAIT_ASYNC_4();
        }
        __syncthreads();

        const unsigned short* wb = smem + WT_OFF + (ks & 1) * WTBUF;
        bf16x16 bh[2], bl[2];
        #pragma unroll
        for (int ni = 0; ni < 2; ++ni) {
            const int off = (ni * 16 + lh) * XROW + ks * 32 + s * 8;
            bh[ni] = load_frag(xhi + off);
            bl[ni] = load_frag(xlo + off);
        }
        #pragma unroll
        for (int mi = 0; mi < 4; ++mi) {
            bf16x16 a = load_frag(wb + (mbase + mi * 16 + lh) * WROW + s * 8);
            #pragma unroll
            for (int ni = 0; ni < 2; ++ni) {
                acc[mi][ni] = __builtin_amdgcn_wmma_f32_16x16x32_bf16(
                    false, a, false, bh[ni], (short)0, acc[mi][ni], false, false);
                acc[mi][ni] = __builtin_amdgcn_wmma_f32_16x16x32_bf16(
                    false, a, false, bl[ni], (short)0, acc[mi][ni], false, false);
            }
        }
        __syncthreads();     // protect buffer being refilled next iteration
    }

    // epilogue: bias + exact-erf GELU, H -> LDS bf16 hi/lo in [t][h]
    {
        const float* bsp = b_sess + (size_t)sid * HID + mbase;
        #pragma unroll
        for (int mi = 0; mi < 4; ++mi) {
            #pragma unroll
            for (int r = 0; r < 8; ++r) {
                const int m    = mi * 16 + r + 8 * s;
                const float bv = bsp[m];
                #pragma unroll
                for (int ni = 0; ni < 2; ++ni) {
                    const int t = ni * 16 + lh;
                    float v = acc[mi][ni][r] + bv;
                    float g = 0.5f * v * (1.0f + erff(v * 0.70710678118654752f));
                    unsigned short hi = f2bf(g);
                    hhi[t * HROW + mbase + m] = hi;
                    hlo[t * HROW + mbase + m] = f2bf(g - bf2f(hi));
                }
            }
        }
    }

    // ---- stage 2: out = W2^T H + b2, tile 256 x NT, double-buffered W2 ----
    f32x8 acc2[2][2];
    {
        const f32x8 z = {0.f, 0.f, 0.f, 0.f, 0.f, 0.f, 0.f, 0.f};
        #pragma unroll
        for (int mi = 0; mi < 2; ++mi)
            #pragma unroll
            for (int ni = 0; ni < 2; ++ni) acc2[mi][ni] = z;
    }
    const int m2base = wave * 32;

    for (int k2 = 0; k2 < 16; ++k2) {
        if (k2 < 15) {
            unsigned short* wb = smem + WT_OFF + ((k2 + 1) & 1) * WTBUF;
            const unsigned short* g0 = Wt2 + (k2 + 1) * 32;
            for (int e = tid; e < NSRC * 4; e += 256) {
                int r = e >> 2, seg = e & 3;
                cp16(g0 + (size_t)r * HID + seg * 8, wb + r * WROW + seg * 8);
            }
            WAIT_ASYNC_4();
        } else {
            WAIT_ASYNC_0();
        }
        __syncthreads();     // k2==0: also orders H writes before H reads

        const unsigned short* wb = smem + WT_OFF + (k2 & 1) * WTBUF;
        bf16x16 bh[2], bl[2];
        #pragma unroll
        for (int ni = 0; ni < 2; ++ni) {
            const int off = (ni * 16 + lh) * HROW + k2 * 32 + s * 8;
            bh[ni] = load_frag(hhi + off);
            bl[ni] = load_frag(hlo + off);
        }
        #pragma unroll
        for (int mi = 0; mi < 2; ++mi) {
            bf16x16 a = load_frag(wb + (m2base + mi * 16 + lh) * WROW + s * 8);
            #pragma unroll
            for (int ni = 0; ni < 2; ++ni) {
                acc2[mi][ni] = __builtin_amdgcn_wmma_f32_16x16x32_bf16(
                    false, a, false, bh[ni], (short)0, acc2[mi][ni], false, false);
                acc2[mi][ni] = __builtin_amdgcn_wmma_f32_16x16x32_bf16(
                    false, a, false, bl[ni], (short)0, acc2[mi][ni], false, false);
            }
        }
        __syncthreads();
    }

    // epilogue: + b2, time mask, fp32 store
    {
        float* ob = out + ((size_t)b * NSRC) * MAXT + t0;
        #pragma unroll
        for (int mi = 0; mi < 2; ++mi) {
            #pragma unroll
            for (int r = 0; r < 8; ++r) {
                const int src  = m2base + mi * 16 + r + 8 * s;
                const float bv = b2[src];
                #pragma unroll
                for (int ni = 0; ni < 2; ++ni) {
                    const int t = ni * 16 + lh;
                    float v = acc2[mi][ni][r] + bv;
                    v = (t0 + t < slen) ? v : 0.0f;
                    ob[(size_t)src * MAXT + t] = v;
                }
            }
        }
    }
}

extern "C" void kernel_launch(void* const* d_in, const int* in_sizes, int n_in,
                              void* d_out, int out_size, void* d_ws, size_t ws_size,
                              hipStream_t stream) {
    const float* x        = (const float*)d_in[0];
    const int*   sess_ids = (const int*)d_in[1];
    const int*   ch_cnt   = (const int*)d_in[2];
    const int*   seq_lens = (const int*)d_in[3];
    const float* W_sess   = (const float*)d_in[4];
    const float* b_sess   = (const float*)d_in[5];
    const float* W2       = (const float*)d_in[6];
    const float* b2       = (const float*)d_in[7];
    float*       out      = (float*)d_out;
    unsigned short* wsb   = (unsigned short*)d_ws;   // needs 2.25 MB scratch

    const int cgrid = (WSESS_ELEMS + W2_ELEMS + 255) / 256;
    convert_weights_kernel<<<cgrid, 256, 0, stream>>>(W_sess, W2, wsb);

    const int grid = 16 * (MAXT / NT);   // 2048 workgroups
    ssp_fused_kernel<<<grid, 256, 0, stream>>>(
        x, sess_ids, ch_cnt, seq_lens, b_sess, b2, wsb, out);
}
